// PillarEncoder_35227321762494
// MI455X (gfx1250) — matrix operand order repaired and checked by reading, not compiled, not run on hardware
//
#include <hip/hip_runtime.h>

typedef __attribute__((ext_vector_type(16))) _Float16 v16h;
typedef __attribute__((ext_vector_type(8)))  float    v8f;
typedef __attribute__((ext_vector_type(4)))  float    f4v;

#define VXc   0.16f
#define VYc   0.16f
#define XOFFc (0.08f + 0.0f)
#define YOFFc (0.08f + (-39.68f))
#define PCR2c (-3.0f)
#define Cc    64
#define Pc    40000
#define Nc    32
#define XGc   432
#define YGc   496
#define NWAVES 8

__device__ __forceinline__ float wsum(float v) {
    #pragma unroll
    for (int m = 16; m >= 1; m >>= 1) v += __shfl_xor(v, m, 32);
    return v;
}
__device__ __forceinline__ float wmaxr(float v) {
    #pragma unroll
    for (int m = 16; m >= 1; m >>= 1) v = fmaxf(v, __shfl_xor(v, m, 32));
    return v;
}
__device__ __forceinline__ float wminr(float v) {
    #pragma unroll
    for (int m = 16; m >= 1; m >>= 1) v = fminf(v, __shfl_xor(v, m, 32));
    return v;
}

// ---------------- canvas zero-fill (dominant HBM cost: ~219 MB) ----------------
__global__ void zero_canvas(f4v* __restrict__ out, int n4) {
    int i = blockIdx.x * blockDim.x + threadIdx.x;
    int stride = gridDim.x * blockDim.x;
    f4v z = 0.0f;
    for (; i < n4; i += stride) __builtin_nontemporal_store(z, &out[i]);
}

// ---------------- fused pillar encoder ----------------
__global__ __launch_bounds__(256) void pillar_encode(
    const float4* __restrict__ pillars,   // P*N float4 (x,y,z,i)
    const float*  __restrict__ Wpoint,    // 64 x 6
    const float*  __restrict__ g1, const float* __restrict__ b1,
    const float*  __restrict__ Wpillar,   // 64 x 8
    const float*  __restrict__ g2, const float* __restrict__ b2,
    const float*  __restrict__ Wpp,       // 64 x 128
    const float*  __restrict__ g3, const float* __restrict__ b3,
    const int*    __restrict__ coors,     // P x 4 (b, ix, iy, 0)
    const int*    __restrict__ npoints,   // P
    float*        __restrict__ out)       // (4, 64, YG, XG)
{
    __shared__ _Float16 sWpp[Cc * 128];                  // 16 KB
    __shared__ _Float16 sLidar[NWAVES][16 * 128];        // 32 KB
    __shared__ float    sFeat[NWAVES][Nc * 8];           // 8 KB (stride 8)
    __shared__ int      sBase[NWAVES][16];

    const int lane = threadIdx.x & 31;
    const int wv   = threadIdx.x >> 5;
    const int lh   = lane >> 4;      // lane half (0/1)
    const int n16  = lane & 15;
    const float rsq = rsqrtf(1.0f + 1e-5f);

    // stage W_pp into LDS as f16 (B-matrix source)
    for (int i = threadIdx.x; i < Cc * 128; i += 256)
        sWpp[i] = (_Float16)Wpp[i];
    __syncthreads();

    const int pillarBase = blockIdx.x * 128 + wv * 16;

    // per-lane channel constants: lane owns channels c0=lane, c1=lane+32
    const int c0 = lane, c1 = lane + 32;
    float wp0[6], wp1[6];
    #pragma unroll
    for (int j = 0; j < 6; ++j) { wp0[j] = Wpoint[c0 * 6 + j]; wp1[j] = Wpoint[c1 * 6 + j]; }
    float wl0[8], wl1[8];
    #pragma unroll
    for (int j = 0; j < 8; ++j) { wl0[j] = Wpillar[c0 * 8 + j]; wl1[j] = Wpillar[c1 * 8 + j]; }
    const float s1a = g1[c0] * rsq, b1a = b1[c0], s1b = g1[c1] * rsq, b1b = b1[c1];
    const float s2a = g2[c0] * rsq, b2a = b2[c0], s2b = g2[c1] * rsq, b2b = b2[c1];

    for (int p = 0; p < 16; ++p) {
        const int gp = pillarBase + p;
        const bool pv = (gp < Pc);
        float4 pt = make_float4(0.f, 0.f, 0.f, 0.f);
        int np = Nc, ib = 0, ixx = 0, iyy = 0;
        if (pv) {
            pt  = pillars[(size_t)gp * Nc + lane];      // coalesced b128
            np  = npoints[gp];
            ib  = coors[gp * 4 + 0];
            ixx = coors[gp * 4 + 1];
            iyy = coors[gp * 4 + 2];
        }
        const float invn = 1.0f / (float)np;
        const float cx = wsum(pt.x) * invn;
        const float cy = wsum(pt.y) * invn;
        const float cz = wsum(pt.z) * invn;
        const bool  vmask = (lane < np);
        const float mk = vmask ? 1.0f : 0.0f;

        float* fr = &sFeat[wv][lane * 8];
        fr[0] = pt.z * mk;
        fr[1] = (pt.x - cx) * mk;
        fr[2] = (pt.y - cy) * mk;
        fr[3] = (pt.z - cz) * mk;
        fr[4] = (pt.x - ((float)ixx * VXc + XOFFc)) * mk;
        fr[5] = (pt.y - ((float)iyy * VYc + YOFFc)) * mk;
        if (lane == 0) sBase[wv][p] = ((ib * Cc) * YGc + iyy) * XGc + ixx;

        // per-z-block stats (block size = 1.0, z in [-3,1))
        const int bid = (int)floorf(pt.z - PCR2c);
        float pf8[8];
        #pragma unroll
        for (int blk = 0; blk < 4; ++blk) {
            const bool in = vmask && (bid == blk);
            float cnt = wsum(in ? 1.0f : 0.0f);
            float zmx = wmaxr(in ? pt.z :  __builtin_inff() * -1.0f);
            float zmn = wminr(in ? pt.z :  __builtin_inff());
            pf8[2 * blk]     = cnt * invn;
            pf8[2 * blk + 1] = (cnt > 0.0f) ? (zmx - zmn) : 0.0f;
        }
        __syncthreads();

        // point MLP (K=6) + BN + ReLU + max over points; masked rows give relu(b1)
        float acc0 = (np < Nc) ? fmaxf(b1a, 0.0f) : 0.0f;
        float acc1 = (np < Nc) ? fmaxf(b1b, 0.0f) : 0.0f;
        for (int n = 0; n < np; ++n) {               // np uniform across wave
            const float* f = &sFeat[wv][n * 8];      // broadcast reads
            const float f0 = f[0], f1 = f[1], f2 = f[2], f3 = f[3], f4 = f[4], f5 = f[5];
            float d0 = f0*wp0[0] + f1*wp0[1] + f2*wp0[2] + f3*wp0[3] + f4*wp0[4] + f5*wp0[5];
            float d1 = f0*wp1[0] + f1*wp1[1] + f2*wp1[2] + f3*wp1[3] + f4*wp1[4] + f5*wp1[5];
            acc0 = fmaxf(acc0, fmaxf(d0 * s1a + b1a, 0.0f));
            acc1 = fmaxf(acc1, fmaxf(d1 * s1b + b1b, 0.0f));
        }

        // pillar-stat GEMM (K=8) + BN + ReLU
        float e0 = 0.f, e1 = 0.f;
        #pragma unroll
        for (int j = 0; j < 8; ++j) { e0 += pf8[j] * wl0[j]; e1 += pf8[j] * wl1[j]; }
        const float pfa = fmaxf(e0 * s2a + b2a, 0.0f);
        const float pfb = fmaxf(e1 * s2b + b2b, 0.0f);

        // lidar row (128 wide) -> LDS f16: [point_feat(64) | pf(64)]
        _Float16* lr = &sLidar[wv][p * 128];
        lr[c0]      = (_Float16)acc0;
        lr[c1]      = (_Float16)acc1;
        lr[64 + c0] = (_Float16)pfa;
        lr[64 + c1] = (_Float16)pfb;
    }
    __syncthreads();

    // ---------------- WMMA: (16 x 128) x (128 x 64), f16 in / f32 acc ----------------
    typedef union { v16h h; unsigned int u[8]; } frag;
    v8f acc[4] = {};

    const unsigned int* lidar32 = (const unsigned int*)&sLidar[wv][0];
    const unsigned int* wpp32   = (const unsigned int*)sWpp;
    const int arow = n16;   // A-matrix row = local pillar index (lanes 0-15 / 16-31 same M)

    #pragma unroll
    for (int ks = 0; ks < 4; ++ks) {                 // K blocks of 32
        frag a;
        #pragma unroll
        for (int j = 0; j < 8; ++j) {                // A 16x32: lanes 0-15 K=0..15, 16-31 K=16..31
            const int kst = 16 * lh + 2 * j + 32 * ks;
            a.u[j] = lidar32[(arow * 128 + kst) >> 1];
        }
        #pragma unroll
        for (int t = 0; t < 4; ++t) {                // N tiles of 16 channels
            frag b;
            const int nn = 16 * t + n16;
            #pragma unroll
            for (int j = 0; j < 8; ++j) {            // B 32x16: lanes 0-15 K=0..15, 16-31 K=16..31
                const int kk = 16 * lh + 2 * j + 32 * ks;
                b.u[j] = wpp32[(nn * 128 + kk) >> 1];
            }
            acc[t] = __builtin_amdgcn_wmma_f32_16x16x32_f16(
                false, a.h, false, b.h, (short)0, acc[t], false, false);
        }
    }

    // epilogue: BN + ReLU, scatter into canvas (b, ch, y, x)
    const int plane = YGc * XGc;
    #pragma unroll
    for (int t = 0; t < 4; ++t) {
        const int ch  = 16 * t + n16;
        const float s3c = g3[ch] * rsq;
        const float b3c = b3[ch];
        #pragma unroll
        for (int r = 0; r < 8; ++r) {                // C/D layout: VGPR r -> M = r + 8*half
            const int mrow = r + 8 * lh;
            const int gp = pillarBase + mrow;
            if (gp < Pc) {
                const float v = fmaxf(acc[t][r] * s3c + b3c, 0.0f);
                out[(size_t)sBase[wv][mrow] + (size_t)ch * plane] = v;
            }
        }
    }
}

extern "C" void kernel_launch(void* const* d_in, const int* in_sizes, int n_in,
                              void* d_out, int out_size, void* d_ws, size_t ws_size,
                              hipStream_t stream) {
    (void)in_sizes; (void)n_in; (void)d_ws; (void)ws_size;
    const float4* pillars = (const float4*)d_in[0];
    const float*  Wpoint  = (const float*)d_in[1];
    const float*  g1      = (const float*)d_in[2];
    const float*  b1      = (const float*)d_in[3];
    const float*  Wpillar = (const float*)d_in[4];
    const float*  g2      = (const float*)d_in[5];
    const float*  b2      = (const float*)d_in[6];
    const float*  Wpp     = (const float*)d_in[7];
    const float*  g3      = (const float*)d_in[8];
    const float*  b3      = (const float*)d_in[9];
    const int*    coors   = (const int*)d_in[10];
    const int*    npts    = (const int*)d_in[11];
    float*        out     = (float*)d_out;

    const int n4 = out_size / 4;                     // out_size divisible by 4
    zero_canvas<<<4096, 256, 0, stream>>>((f4v*)d_out, n4);

    const int blocks = (Pc + 127) / 128;             // 313, 8 waves x 16 pillars each
    pillar_encode<<<blocks, 256, 0, stream>>>(pillars, Wpoint, g1, b1,
                                              Wpillar, g2, b2, Wpp, g3, b3,
                                              coors, npts, out);
}
